// PointCrossAttention_4157528342754
// MI455X (gfx1250) — matrix operand order repaired and checked by reading, not compiled, not run on hardware
//
#include <hip/hip_runtime.h>
#include <hip/hip_bf16.h>

// ---------------------------------------------------------------------------
// Types for CDNA5 WMMA (wave32, 16x16x32 bf16 -> f32)
// ---------------------------------------------------------------------------
typedef __bf16 bf16;
typedef bf16  v8bf  __attribute__((ext_vector_type(8)));
typedef bf16  v16bf __attribute__((ext_vector_type(16)));
typedef float v8f   __attribute__((ext_vector_type(8)));

constexpr int B_    = 2;
constexpr int PCSZ  = 16384;
constexpr int NPTS  = 4096;   // INPUT_PC
constexpr int NLAT  = 1024;
constexpr int WID   = 1024;
constexpr int HEADS = 16;
constexpr int HD    = 64;
constexpr int KPAD  = 64;     // padded IN_DIM (54 -> 64)
constexpr int INDIM = 54;

static __device__ __forceinline__ v8bf ld8(const bf16* p) {
  return *reinterpret_cast<const v8bf*>(p);
}
static __device__ __forceinline__ v16bf cat8(v8bf lo, v8bf hi) {
  return __builtin_shufflevector(lo, hi, 0,1,2,3,4,5,6,7,8,9,10,11,12,13,14,15);
}
static __device__ __forceinline__ v8f wmma_bf16(v16bf a, v16bf b, v8f c) {
  return __builtin_amdgcn_wmma_f32_16x16x32_bf16(false, a, false, b, (short)0, c, false, false);
}
// reductions within 16-lane half-wave (rows of a C-fragment live across 16 lanes)
static __device__ __forceinline__ float red16_max(float v) {
#pragma unroll
  for (int m = 1; m < 16; m <<= 1) v = fmaxf(v, __shfl_xor(v, m, 32));
  return v;
}
static __device__ __forceinline__ float red16_sum(float v) {
#pragma unroll
  for (int m = 1; m < 16; m <<= 1) v += __shfl_xor(v, m, 32);
  return v;
}

// ---------------------------------------------------------------------------
// Gather permuted points/features
// ---------------------------------------------------------------------------
__global__ void k_gather(const float* __restrict__ pc, const float* __restrict__ ft,
                         const int* __restrict__ perm,
                         float* __restrict__ gp, float* __restrict__ gf) {
  int i = blockIdx.x * blockDim.x + threadIdx.x;
  if (i >= B_ * NPTS) return;
  int b = i / NPTS, t = i - b * NPTS;
  int s = perm[t];
  size_t src = ((size_t)b * PCSZ + s) * 3;
#pragma unroll
  for (int c = 0; c < 3; ++c) {
    gp[(size_t)i * 3 + c] = pc[src + c];
    gf[(size_t)i * 3 + c] = ft[src + c];
  }
}

// ---------------------------------------------------------------------------
// Farthest point sampling: one block per batch, 1023 dependent argmax steps.
// Wave-level shfl argmax + tiny LDS merge -> 3 barriers per step.
// ---------------------------------------------------------------------------
__global__ __launch_bounds__(256) void k_fps(const float* __restrict__ pts, int* __restrict__ out) {
  int b = blockIdx.x;
  const float* P = pts + (size_t)b * NPTS * 3;
  __shared__ float dists[NPTS];
  __shared__ float wr_v[8];
  __shared__ int   wr_i[8];
  __shared__ float lp[3];
  __shared__ int   s_last;
  int tid = threadIdx.x;
  for (int i = tid; i < NPTS; i += 256) dists[i] = 1e10f;
  if (tid == 0) { s_last = 0; out[b * NLAT] = 0; }
  __syncthreads();
  for (int it = 1; it < NLAT; ++it) {
    if (tid < 3) lp[tid] = P[(size_t)s_last * 3 + tid];
    __syncthreads();
    float bx = lp[0], by = lp[1], bz = lp[2];
    float bm = -1.f; int bi = 0;
    for (int i = tid; i < NPTS; i += 256) {
      float dx = P[i * 3 + 0] - bx;
      float dy = P[i * 3 + 1] - by;
      float dz = P[i * 3 + 2] - bz;
      float d  = dx * dx + dy * dy + dz * dz;
      float nd = fminf(dists[i], d);
      dists[i] = nd;
      if (nd > bm) { bm = nd; bi = i; }
    }
#pragma unroll
    for (int m = 1; m < 32; m <<= 1) {
      float ov = __shfl_xor(bm, m, 32);
      int   oi = __shfl_xor(bi, m, 32);
      if (ov > bm || (ov == bm && oi < bi)) { bm = ov; bi = oi; }
    }
    if ((tid & 31) == 0) { wr_v[tid >> 5] = bm; wr_i[tid >> 5] = bi; }
    __syncthreads();
    if (tid == 0) {
      float v = wr_v[0]; int ix = wr_i[0];
      for (int w = 1; w < 8; ++w)
        if (wr_v[w] > v || (wr_v[w] == v && wr_i[w] < ix)) { v = wr_v[w]; ix = wr_i[w]; }
      s_last = ix; out[b * NLAT + it] = ix;
    }
    __syncthreads();
  }
}

// ---------------------------------------------------------------------------
// Fourier embed -> bf16 [tok][64] (cols 54..63 zero-padded).
// layout: [xyz(3) | sin(coord*2^f) 24 | cos 24 | feats 3]
// ---------------------------------------------------------------------------
__global__ void k_embed(const float* __restrict__ gp, const float* __restrict__ gf,
                        const int* __restrict__ qidx, bf16* __restrict__ out, int T) {
  int tok = blockIdx.x;
  int c = threadIdx.x;                 // 0..63
  int b = tok / T, t = tok - b * T;
  int row = qidx ? qidx[b * NLAT + t] : t;
  const float* p = gp + ((size_t)b * NPTS + row) * 3;
  const float* f = gf + ((size_t)b * NPTS + row) * 3;
  float v = 0.f;
  if (c < 3)        v = p[c];
  else if (c < 27)  { int j = c - 3;  v = sinf(p[j >> 3] * (float)(1 << (j & 7))); }
  else if (c < 51)  { int j = c - 27; v = cosf(p[j >> 3] * (float)(1 << (j & 7))); }
  else if (c < INDIM) v = f[c - 51];
  out[(size_t)tok * KPAD + c] = (bf16)v;
}

// ---------------------------------------------------------------------------
// Weight convert+transpose: f32 [K][N] -> bf16 [N][Kp] (zero pad K..Kp)
// Makes WMMA B-fragments contiguous 16-half chunks.
// ---------------------------------------------------------------------------
__global__ void k_convert_w(const float* __restrict__ src, bf16* __restrict__ dst,
                            int K, int N, int Kp) {
  int i = blockIdx.x * 256 + threadIdx.x;
  if (i >= N * Kp) return;
  int n = i / Kp, k = i - n * Kp;
  dst[i] = (k < K) ? (bf16)src[(size_t)k * N + n] : (bf16)0.f;
}

// ---------------------------------------------------------------------------
// GEMM: C[M][N] f32 = A[M][K] bf16 x Wt[N][K] bf16 (+bias).
// 256 threads = 8 waves; block tile 128x128; wave tile 32x64 (8 accumulators).
// Ping-pong double-buffered K loop (unroll x2, K always a multiple of 64):
// the two fragment sets alternate by NAME, so no register-rotation copies
// (the v_dual_mov chains seen previously) appear; loads for chunk k+32 issue
// before the WMMAs of chunk k, overlapping global_load latency with the
// matrix cores.
// Fragment layouts used here (A and B share contiguous k = 16h+s order):
//   A (16x32): lane L -> row L&15; 16 contiguous K halves at kb=(L>>4)*16
//   B (32x16): lane L -> col L&15; 16 contiguous K halves at kb=(L>>4)*16
//   C (16x16): lane L -> col L&15; rows (L<16?0:8)+r for vreg r
// ---------------------------------------------------------------------------
struct Frags { v16bf a[2]; v16bf b[4]; };

__global__ __launch_bounds__(256) void k_gemm(const bf16* __restrict__ A,
                                              const bf16* __restrict__ Wt,
                                              const float* __restrict__ bias,
                                              float* __restrict__ C,
                                              int M, int N, int K) {
  int wv = threadIdx.x >> 5, lane = threadIdx.x & 31;
  int l15 = lane & 15, hi = lane >> 4;
  int wm = blockIdx.y * 128 + (wv & 3) * 32;
  int wn = blockIdx.x * 128 + (wv >> 2) * 64;
  const bf16* a0 = A + (size_t)(wm + l15) * K + hi * 16;
  const bf16* a1 = A + (size_t)(wm + 16 + l15) * K + hi * 16;
  const bf16* bp0 = Wt + (size_t)(wn + 0 * 16 + l15) * K + hi * 16;
  const bf16* bp1 = Wt + (size_t)(wn + 1 * 16 + l15) * K + hi * 16;
  const bf16* bp2 = Wt + (size_t)(wn + 2 * 16 + l15) * K + hi * 16;
  const bf16* bp3 = Wt + (size_t)(wn + 3 * 16 + l15) * K + hi * 16;

  v8f acc[2][4] = {};

  auto loadf = [&](int kc, Frags& f) {
    f.a[0] = cat8(ld8(a0 + kc), ld8(a0 + kc + 8));
    f.a[1] = cat8(ld8(a1 + kc), ld8(a1 + kc + 8));
    f.b[0] = cat8(ld8(bp0 + kc), ld8(bp0 + kc + 8));
    f.b[1] = cat8(ld8(bp1 + kc), ld8(bp1 + kc + 8));
    f.b[2] = cat8(ld8(bp2 + kc), ld8(bp2 + kc + 8));
    f.b[3] = cat8(ld8(bp3 + kc), ld8(bp3 + kc + 8));
  };
  auto compute = [&](const Frags& f) {
#pragma unroll
    for (int s = 0; s < 2; ++s)
#pragma unroll
      for (int t = 0; t < 4; ++t)
        acc[s][t] = wmma_bf16(f.a[s], f.b[t], acc[s][t]);
  };

  Frags f0, f1;
  loadf(0, f0);
  int kc = 0;
  for (; kc + 64 < K; kc += 64) {
    __builtin_prefetch(a0 + kc + 256, 0, 1);   // global_prefetch_b8, L2 warm-up
    loadf(kc + 32, f1);
    compute(f0);
    loadf(kc + 64, f0);
    compute(f1);
  }
  // tail: last two 32-chunks (K is a multiple of 64)
  loadf(kc + 32, f1);
  compute(f0);
  compute(f1);

#pragma unroll
  for (int t = 0; t < 4; ++t) {
    int n = wn + t * 16 + l15;
    float bv = bias ? bias[n] : 0.f;
#pragma unroll
    for (int s = 0; s < 2; ++s) {
      int mrow = wm + s * 16 + hi * 8;
#pragma unroll
      for (int r = 0; r < 8; ++r)
        C[(size_t)(mrow + r) * N + n] = acc[s][t][r] + bv;
    }
  }
}

// ---------------------------------------------------------------------------
// LayerNorm over width 1024; writes bf16 (GEMM input) and/or f32 (output).
// ---------------------------------------------------------------------------
__global__ __launch_bounds__(256) void k_ln(const float* __restrict__ x,
                                            const float* __restrict__ g,
                                            const float* __restrict__ bb,
                                            bf16* __restrict__ ob, float* __restrict__ of,
                                            float eps) {
  int t = blockIdx.x;
  const float* row = x + (size_t)t * WID;
  __shared__ float red[256];
  __shared__ float s_mu, s_rs;
  float s = 0.f;
  for (int i = threadIdx.x; i < WID; i += 256) s += row[i];
  red[threadIdx.x] = s; __syncthreads();
  for (int k = 128; k > 0; k >>= 1) { if (threadIdx.x < k) red[threadIdx.x] += red[threadIdx.x + k]; __syncthreads(); }
  if (threadIdx.x == 0) s_mu = red[0] * (1.f / WID);
  __syncthreads();
  float mu = s_mu, v = 0.f;
  for (int i = threadIdx.x; i < WID; i += 256) { float d = row[i] - mu; v += d * d; }
  red[threadIdx.x] = v; __syncthreads();
  for (int k = 128; k > 0; k >>= 1) { if (threadIdx.x < k) red[threadIdx.x] += red[threadIdx.x + k]; __syncthreads(); }
  if (threadIdx.x == 0) s_rs = rsqrtf(red[0] * (1.f / WID) + eps);
  __syncthreads();
  float rs = s_rs;
  for (int i = threadIdx.x; i < WID; i += 256) {
    float y = (row[i] - mu) * rs * g[i] + bb[i];
    if (ob) ob[(size_t)t * WID + i] = (bf16)y;
    if (of) of[(size_t)t * WID + i] = y;
  }
}

// ---------------------------------------------------------------------------
// Per-head pack: optional LN over head_dim (64, eps 1e-6), bf16 output.
// dst normal: [b*H+h][T][64]; transposed (for V): [b*H+h][64][T]
// One wave per (b,h,t): lane handles dims lane, lane+32.
// ---------------------------------------------------------------------------
__global__ __launch_bounds__(32) void k_head_pack(const float* __restrict__ src,
                                                  int tokStride, int headMult, int off,
                                                  const float* __restrict__ g,
                                                  const float* __restrict__ bb,
                                                  bf16* __restrict__ dst, int T, int transposed) {
  int t = blockIdx.x, h = blockIdx.y, b = blockIdx.z;
  int lane = threadIdx.x;
  const float* row = src + ((size_t)(b * T + t)) * tokStride + h * headMult + off;
  float x0 = row[lane], x1 = row[lane + 32];
  if (g) {
    float s = x0 + x1;
#pragma unroll
    for (int m = 1; m < 32; m <<= 1) s += __shfl_xor(s, m, 32);
    float mu = s * (1.f / 64.f);
    float d0 = x0 - mu, d1 = x1 - mu;
    float v = d0 * d0 + d1 * d1;
#pragma unroll
    for (int m = 1; m < 32; m <<= 1) v += __shfl_xor(v, m, 32);
    float rs = rsqrtf(v * (1.f / 64.f) + 1e-6f);
    x0 = d0 * rs * g[lane] + bb[lane];
    x1 = d1 * rs * g[lane + 32] + bb[lane + 32];
  }
  size_t bh = (size_t)(b * HEADS + h);
  if (!transposed) {
    bf16* d = dst + (bh * T + t) * HD;
    d[lane] = (bf16)x0; d[lane + 32] = (bf16)x1;
  } else {
    bf16* d = dst + bh * HD * (size_t)T;
    d[(size_t)lane * T + t] = (bf16)x0;
    d[(size_t)(lane + 32) * T + t] = (bf16)x1;
  }
}

// ---------------------------------------------------------------------------
// Flash attention: 1 wave = 16 queries x head_dim 64; online softmax.
// Scores: S(16x16) = sum_c wmma(Q[:,32c..], K^T chunk). P staged via LDS to
// re-layout C-frag -> A-frag (s_wait_dscnt 0; per-wave LDS region, no barrier
// needed), then O += wmma(P, V^T-frag) x4.
// ---------------------------------------------------------------------------
__global__ __launch_bounds__(128) void k_attn(const bf16* __restrict__ Qh,
                                              const bf16* __restrict__ Kh,
                                              const bf16* __restrict__ Vt,
                                              float* __restrict__ out, int Tq, int Tk) {
  int wv = threadIdx.x >> 5, lane = threadIdx.x & 31;
  int h = blockIdx.y, b = blockIdx.z;
  int qt = blockIdx.x * 4 + wv;
  int l15 = lane & 15, hi = lane >> 4;
  size_t bh = (size_t)(b * HEADS + h);
  const bf16* Qb = Qh + bh * Tq * HD;
  const bf16* Kb = Kh + bh * Tk * HD;
  const bf16* Vb = Vt + bh * HD * (size_t)Tk;

  v16bf qa[2];
#pragma unroll
  for (int c = 0; c < 2; ++c) {
    const bf16* p = Qb + (size_t)(qt * 16 + l15) * HD + c * 32 + hi * 16;
    qa[c] = cat8(ld8(p), ld8(p + 8));
  }
  v8f acc[4] = {};
  float m[8], l[8];
#pragma unroll
  for (int r = 0; r < 8; ++r) { m[r] = -1e30f; l[r] = 0.f; }

  __shared__ bf16 plds[4][16][32];
  bf16 (*pl)[32] = plds[wv];

  for (int kt = 0; kt < Tk; kt += 32) {
    v8f sc[2] = {};
#pragma unroll
    for (int s = 0; s < 2; ++s) {
      const bf16* kp = Kb + (size_t)(kt + s * 16 + l15) * HD + hi * 16;
#pragma unroll
      for (int c = 0; c < 2; ++c) {
        v16bf kb = cat8(ld8(kp + c * 32), ld8(kp + c * 32 + 8));
        sc[s] = wmma_bf16(qa[c], kb, sc[s]);
      }
    }
#pragma unroll
    for (int r = 0; r < 8; ++r) {
      float v0 = sc[0][r] * 0.125f;     // 1/sqrt(64)
      float v1 = sc[1][r] * 0.125f;
      float rm = red16_max(fmaxf(v0, v1));
      float mn = fmaxf(m[r], rm);
      float alpha = __expf(m[r] - mn);
      m[r] = mn;
      float p0 = __expf(v0 - mn);
      float p1 = __expf(v1 - mn);
      l[r] = l[r] * alpha + red16_sum(p0 + p1);
#pragma unroll
      for (int t = 0; t < 4; ++t) acc[t][r] = acc[t][r] * alpha;
      pl[hi * 8 + r][l15]      = (bf16)p0;
      pl[hi * 8 + r][16 + l15] = (bf16)p1;
    }
    asm volatile("s_wait_dscnt 0" ::: "memory");
    v16bf pa = cat8(*reinterpret_cast<const v8bf*>(&pl[l15][hi * 16]),
                    *reinterpret_cast<const v8bf*>(&pl[l15][hi * 16 + 8]));
#pragma unroll
    for (int t = 0; t < 4; ++t) {
      const bf16* vp = Vb + (size_t)(t * 16 + l15) * Tk + kt + hi * 16;
      v16bf vb = cat8(ld8(vp), ld8(vp + 8));
      acc[t] = wmma_bf16(pa, vb, acc[t]);
    }
  }
  float* ob = out + (size_t)b * Tq * WID;
#pragma unroll
  for (int t = 0; t < 4; ++t)
#pragma unroll
    for (int r = 0; r < 8; ++r) {
      int q = qt * 16 + hi * 8 + r;
      ob[(size_t)q * WID + h * HD + t * 16 + l15] = acc[t][r] / l[r];
    }
}

// ---------------------------------------------------------------------------
// Elementwise helpers
// ---------------------------------------------------------------------------
__global__ void k_addo(float* __restrict__ o, const float* __restrict__ a,
                       const float* __restrict__ b, int n) {
  int i = blockIdx.x * 256 + threadIdx.x;
  if (i < n) o[i] = a[i] + b[i];
}
__global__ void k_addi(float* __restrict__ o, const float* __restrict__ a, int n) {
  int i = blockIdx.x * 256 + threadIdx.x;
  if (i < n) o[i] += a[i];
}
__global__ void k_f2b(const float* __restrict__ a, bf16* __restrict__ o, int n) {
  int i = blockIdx.x * 256 + threadIdx.x;
  if (i < n) o[i] = (bf16)a[i];
}
__global__ void k_gelu(const float* __restrict__ a, bf16* __restrict__ o, int n) {
  int i = blockIdx.x * 256 + threadIdx.x;
  if (i < n) {
    float x = a[i];
    o[i] = (bf16)(0.5f * x * (1.f + erff(x * 0.7071067811865475f)));
  }
}

// ---------------------------------------------------------------------------
// Host orchestration
// ---------------------------------------------------------------------------
extern "C" void kernel_launch(void* const* d_in, const int* in_sizes, int n_in,
                              void* d_out, int out_size, void* d_ws, size_t ws_size,
                              hipStream_t stream) {
  (void)in_sizes; (void)n_in; (void)out_size; (void)ws_size;
  auto F = [&](int i) { return (const float*)d_in[i]; };
  const float* pc   = F(0);
  const float* ftr  = F(1);
  const int*   perm = (const int*)d_in[2];
  const float* ip_w = F(3);
  const float* ip_b = F(4);
  auto CF = [&](int j) { return F(5 + j); };           // cross params
  auto LF = [&](int l, int j) { return F(23 + 15 * l + j); };  // layer params

  // ---- workspace bump allocator ----
  char* wsp = (char*)d_ws; size_t off = 0;
  auto alloc = [&](size_t bytes) -> void* {
    void* p = wsp + off; off += (bytes + 255) & ~(size_t)255; return p;
  };
  bf16*  wbuf   = (bf16*) alloc((size_t)4096 * 1024 * sizeof(bf16));  // max weight (8MB)
  float* gp     = (float*)alloc((size_t)B_ * NPTS * 3 * 4);
  float* gf     = (float*)alloc((size_t)B_ * NPTS * 3 * 4);
  int*   qidx   = (int*)  alloc((size_t)B_ * NLAT * 4);
  bf16*  embq   = (bf16*) alloc((size_t)B_ * NLAT * KPAD * 2);
  bf16*  embd   = (bf16*) alloc((size_t)B_ * NPTS * KPAD * 2);
  float* qlat   = (float*)alloc((size_t)B_ * NLAT * WID * 4);
  float* dtok   = (float*)alloc((size_t)B_ * NPTS * WID * 4);
  bf16*  xln    = (bf16*) alloc((size_t)B_ * NLAT * WID * 2);        // LN out / attn-out bf16
  bf16*  dln    = (bf16*) alloc((size_t)B_ * NPTS * WID * 2);        // data LN; reused: MLP hidden bf16
  float* bigf   = (float*)alloc((size_t)B_ * NPTS * 2 * WID * 4);    // kv / qkv / MLP hidden f32
  bf16*  Qh     = (bf16*) alloc((size_t)B_ * HEADS * NLAT * HD * 2);
  bf16*  Kh     = (bf16*) alloc((size_t)B_ * HEADS * NPTS * HD * 2);
  bf16*  Vt     = (bf16*) alloc((size_t)B_ * HEADS * NPTS * HD * 2);
  float* atto   = (float*)alloc((size_t)B_ * NLAT * WID * 4);
  float* xres   = (float*)alloc((size_t)B_ * NLAT * WID * 4);
  float* tmpf   = (float*)alloc((size_t)B_ * NLAT * WID * 4);

  const int TL = B_ * NLAT;     // 2048 latent tokens (batches fused)
  const int TD = B_ * NPTS;     // 8192 data tokens
  const int nL = TL * WID;

  auto convw = [&](const float* src, int K, int N, int Kp) {
    int n = N * Kp;
    k_convert_w<<<(n + 255) / 256, 256, 0, stream>>>(src, wbuf, K, N, Kp);
  };
  auto gemm = [&](const bf16* A, const float* bias, float* Cout, int M, int N, int K) {
    k_gemm<<<dim3(N / 128, M / 128), 256, 0, stream>>>(A, wbuf, bias, Cout, M, N, K);
  };
  auto ln = [&](const float* x, const float* g, const float* bb, bf16* ob, float* of2,
                float eps, int T) {
    k_ln<<<T, 256, 0, stream>>>(x, g, bb, ob, of2, eps);
  };
  auto add_in = [&](float* o, const float* a, int n) {
    k_addi<<<(n + 255) / 256, 256, 0, stream>>>(o, a, n);
  };

  // ---- preprocessing ----
  k_gather<<<(B_ * NPTS + 255) / 256, 256, 0, stream>>>(pc, ftr, perm, gp, gf);
  k_fps<<<B_, 256, 0, stream>>>(gp, qidx);
  k_embed<<<B_ * NPTS, 64, 0, stream>>>(gp, gf, nullptr, embd, NPTS);
  k_embed<<<B_ * NLAT, 64, 0, stream>>>(gp, gf, qidx, embq, NLAT);

  // ---- input projection (54 -> 1024, K padded to 64) ----
  convw(ip_w, INDIM, WID, KPAD);
  gemm(embq, ip_b, qlat, TL, WID, KPAD);
  gemm(embd, ip_b, dtok, TD, WID, KPAD);

  // ---- cross attention block ----
  ln(qlat, CF(0), CF(1), xln, nullptr, 1e-6f, TL);      // ln1(query)
  ln(dtok, CF(2), CF(3), dln, nullptr, 1e-6f, TD);      // ln2(data)
  convw(CF(6), WID, WID, WID);                          // c_q_w
  gemm(xln, nullptr, tmpf, TL, WID, WID);               // q
  convw(CF(7), WID, 2 * WID, WID);                      // c_kv_w
  gemm(dln, nullptr, bigf, TD, 2 * WID, WID);           // kv
  k_head_pack<<<dim3(NLAT, HEADS, B_), 32, 0, stream>>>(tmpf, WID, HD, 0, CF(10), CF(11), Qh, NLAT, 0);
  k_head_pack<<<dim3(NPTS, HEADS, B_), 32, 0, stream>>>(bigf, 2 * WID, 2 * HD, 0, CF(12), CF(13), Kh, NPTS, 0);
  k_head_pack<<<dim3(NPTS, HEADS, B_), 32, 0, stream>>>(bigf, 2 * WID, 2 * HD, HD, nullptr, nullptr, Vt, NPTS, 1);
  k_attn<<<dim3(NLAT / 64, HEADS, B_), 128, 0, stream>>>(Qh, Kh, Vt, atto, NLAT, NPTS);
  k_f2b<<<(nL + 255) / 256, 256, 0, stream>>>(atto, xln, nL);
  convw(CF(8), WID, WID, WID);                          // c_proj_w
  gemm(xln, CF(9), tmpf, TL, WID, WID);
  k_addo<<<(nL + 255) / 256, 256, 0, stream>>>(xres, qlat, tmpf, nL);  // x = query + attn

  // cross MLP
  ln(xres, CF(4), CF(5), xln, nullptr, 1e-6f, TL);      // ln3
  convw(CF(14), WID, 4 * WID, WID);                     // fc_w
  gemm(xln, CF(15), bigf, TL, 4 * WID, WID);
  k_gelu<<<(TL * 4 * WID + 255) / 256, 256, 0, stream>>>(bigf, dln, TL * 4 * WID);
  convw(CF(16), 4 * WID, WID, 4 * WID);                 // pj_w
  gemm(dln, CF(17), tmpf, TL, WID, 4 * WID);
  add_in(xres, tmpf, nL);

  // ---- self-attention layers ----
  for (int lyr = 0; lyr < 8; ++lyr) {
    ln(xres, LF(lyr, 0), LF(lyr, 1), xln, nullptr, 1e-6f, TL);
    convw(LF(lyr, 4), WID, 3 * WID, WID);               // c_qkv_w
    gemm(xln, nullptr, bigf, TL, 3 * WID, WID);
    k_head_pack<<<dim3(NLAT, HEADS, B_), 32, 0, stream>>>(bigf, 3 * WID, 3 * HD, 0,      LF(lyr, 7), LF(lyr, 8),  Qh, NLAT, 0);
    k_head_pack<<<dim3(NLAT, HEADS, B_), 32, 0, stream>>>(bigf, 3 * WID, 3 * HD, HD,     LF(lyr, 9), LF(lyr, 10), Kh, NLAT, 0);
    k_head_pack<<<dim3(NLAT, HEADS, B_), 32, 0, stream>>>(bigf, 3 * WID, 3 * HD, 2 * HD, nullptr,    nullptr,     Vt, NLAT, 1);
    k_attn<<<dim3(NLAT / 64, HEADS, B_), 128, 0, stream>>>(Qh, Kh, Vt, atto, NLAT, NLAT);
    k_f2b<<<(nL + 255) / 256, 256, 0, stream>>>(atto, xln, nL);
    convw(LF(lyr, 5), WID, WID, WID);                   // c_proj_w
    gemm(xln, LF(lyr, 6), tmpf, TL, WID, WID);
    add_in(xres, tmpf, nL);

    ln(xres, LF(lyr, 2), LF(lyr, 3), xln, nullptr, 1e-6f, TL);
    convw(LF(lyr, 11), WID, 4 * WID, WID);              // fc_w
    gemm(xln, LF(lyr, 12), bigf, TL, 4 * WID, WID);
    k_gelu<<<(TL * 4 * WID + 255) / 256, 256, 0, stream>>>(bigf, dln, TL * 4 * WID);
    convw(LF(lyr, 13), 4 * WID, WID, 4 * WID);          // pj_w
    gemm(dln, LF(lyr, 14), tmpf, TL, WID, 4 * WID);
    add_in(xres, tmpf, nL);
  }

  // ---- final LN (eps 1e-5) straight to output ----
  ln(xres, F(143), F(144), nullptr, (float*)d_out, 1e-5f, TL);
}
